// NonLocalNd_bn_cbam_31198642438286
// MI455X (gfx1250) — compile-verified, matrix operand order checked by reading, not run on hardware
//
#include <hip/hip_runtime.h>
#include <hip/hip_bf16.h>
#include <math.h>

typedef __bf16 bf16_t;
typedef __bf16 v16bf __attribute__((ext_vector_type(16)));
typedef float  v8f   __attribute__((ext_vector_type(8)));

#define DEV static __device__ __forceinline__

// ---- WMMA fragment helpers (CDNA5 16x16x32 bf16 layouts, ISA 7.12.2) ----
// A 16x32 (rows contiguous in LDS, stride in elements). Lane l: M = l&15,
// h = (l>>4)*8; elems 0..7 = row[h..h+7], elems 8..15 = row[16+h..16+h+7].
// B 32x16 loads identically when stored as [N][K] (column-major columns).
DEV v16bf load_frag(const bf16_t* base, int stride, int lane) {
  int m = lane & 15;
  int h = (lane >> 4) << 3;
  const bf16_t* r = base + m * stride + h;
  v16bf a;
#pragma unroll
  for (int i = 0; i < 8; ++i) a[i] = r[i];
#pragma unroll
  for (int i = 0; i < 8; ++i) a[8 + i] = r[16 + i];
  return a;
}

DEV v8f wmma_bf16(v16bf a, v16bf b, v8f c) {
  return __builtin_amdgcn_wmma_f32_16x16x32_bf16(false, a, false, b,
                                                 (short)0, c, false, false);
}

// ---- CDNA5 async global->LDS copy (ASYNCcnt path, no VGPR round-trip) ----
// Aperture rule (ISA 10.2): flat-addr low 32 bits == LDS byte offset, so the
// truncated pointer is the VDST (LDS address) operand.
DEV void cp_async_b128(bf16_t* lds_dst, const bf16_t* gsrc) {
  asm volatile("global_load_async_to_lds_b128 %0, %1, off"
               :: "v"((unsigned)(uintptr_t)lds_dst), "v"(gsrc)
               : "memory");
}
DEV void async_wait0() {
  asm volatile("s_wait_asynccnt 0x0" ::: "memory");
}

// ------------------------------ small kernels ------------------------------
__global__ void k_cast(const float* __restrict__ s, bf16_t* __restrict__ d, int n) {
  int i = blockIdx.x * 256 + threadIdx.x;
  if (i < n) d[i] = (bf16_t)s[i];
}

// maxpool 2x2: x[B][256][64][64] -> xp[B][256][1024]
__global__ void k_pool(const float* __restrict__ x, float* __restrict__ xp) {
  size_t i = (size_t)blockIdx.x * 256 + threadIdx.x;
  if (i >= (size_t)8 * 256 * 1024) return;
  int m = (int)(i & 1023);
  size_t bc = i >> 10;
  int hp = m >> 5, wp = m & 31;
  const float* s = x + (bc << 12) + (size_t)(hp * 2) * 64 + wp * 2;
  xp[i] = fmaxf(fmaxf(s[0], s[1]), fmaxf(s[64], s[65]));
}

// [B][C][N] f32 -> [B][N][C] bf16 via LDS tile transpose
__global__ void k_transpose_cast(const float* __restrict__ src,
                                 bf16_t* __restrict__ dst, int C, int N) {
  __shared__ float tile[32][33];
  int b = blockIdx.z;
  int n0 = blockIdx.x * 32, c0 = blockIdx.y * 32;
  int tx = threadIdx.x, ty = threadIdx.y;  // 32 x 8
  const float* sp = src + (size_t)b * C * N;
  bf16_t* dp = dst + (size_t)b * N * C;
#pragma unroll
  for (int i = 0; i < 4; ++i)
    tile[ty + 8 * i][tx] = sp[(size_t)(c0 + ty + 8 * i) * N + n0 + tx];
  __syncthreads();
#pragma unroll
  for (int i = 0; i < 4; ++i)
    dp[(size_t)(n0 + ty + 8 * i) * C + c0 + tx] = (bf16_t)tile[tx][ty + 8 * i];
}

// ------------------------ projection GEMM (WMMA) ---------------------------
// D[64 n][128 p] per workgroup.  mode 0/1: q/k (spatial whiten + BN stats),
// mode 2: v (write [B][128][N] fp32 + bf16).
__global__ void k_proj(const bf16_t* __restrict__ src_t,  // [B][N][256]
                       const bf16_t* __restrict__ wt,     // [128][256]
                       const float* __restrict__ bias,    // [128]
                       bf16_t* __restrict__ out_cent,     // [B][N][128]
                       bf16_t* __restrict__ v_bf,         // [B][128][N]
                       float* __restrict__ v_f32,         // [B][128][N]
                       float* __restrict__ ch_sum, float* __restrict__ ch_sumsq,
                       int N, int mode) {
  extern __shared__ char smem_dyn[];
  const int AS = 72, WS = 72, SS = 132;
  bf16_t* Ablk = (bf16_t*)smem_dyn;            // [64][72]
  bf16_t* Wblk = Ablk + 64 * AS;               // [128][72]
  float*  Sl   = (float*)(Wblk + 128 * WS);    // [64][132]
  float*  rmean = Sl + 64 * SS;                // [64]
  int tid = threadIdx.x, lane = tid & 31, wid = tid >> 5;
  int b = blockIdx.y, n0 = blockIdx.x * 64;
  int m0 = (wid >> 1) * 16, cg = wid & 1;
  int l15 = lane & 15, hi8 = (lane >> 4) * 8;
  v8f acc[4] = {};
  const bf16_t* srow = src_t + ((size_t)b * N + n0) * 256;
  for (int kc = 0; kc < 256; kc += 64) {
    for (int i = tid; i < 64 * 8; i += 256) {         // A: 64 rows x 8 x 16B
      int r = i >> 3, s = i & 7;
      cp_async_b128(&Ablk[r * AS + s * 8], srow + (size_t)r * 256 + kc + s * 8);
    }
    for (int i = tid; i < 128 * 8; i += 256) {        // W: 128 rows x 8 x 16B
      int r = i >> 3, s = i & 7;
      cp_async_b128(&Wblk[r * WS + s * 8], wt + (size_t)r * 256 + kc + s * 8);
    }
    async_wait0();
    __syncthreads();
#pragma unroll
    for (int kk = 0; kk < 64; kk += 32) {
      v16bf a = load_frag(&Ablk[m0 * AS + kk], AS, lane);
#pragma unroll
      for (int t = 0; t < 4; ++t) {
        v16bf bm = load_frag(&Wblk[(cg * 64 + t * 16) * WS + kk], WS, lane);
        acc[t] = wmma_bf16(a, bm, acc[t]);
      }
    }
    __syncthreads();
  }
#pragma unroll
  for (int t = 0; t < 4; ++t) {
    float bb = bias[cg * 64 + t * 16 + l15];
#pragma unroll
    for (int r = 0; r < 8; ++r) acc[t][r] += bb;
  }
  if (mode == 2) {
#pragma unroll
    for (int t = 0; t < 4; ++t) {
      int p = cg * 64 + t * 16 + l15;
#pragma unroll
      for (int r = 0; r < 8; ++r) {
        size_t o = ((size_t)b * 128 + p) * N + (n0 + m0 + r + hi8);
        v_f32[o] = acc[t][r];
        v_bf[o] = (bf16_t)acc[t][r];
      }
    }
    return;
  }
#pragma unroll
  for (int t = 0; t < 4; ++t) {
    int c = cg * 64 + t * 16 + l15;
#pragma unroll
    for (int r = 0; r < 8; ++r) Sl[(m0 + r + hi8) * SS + c] = acc[t][r];
  }
  __syncthreads();
  if (tid < 64) {  // per-position channel mean (spatial whitening)
    float s = 0.f;
    for (int c = 0; c < 128; ++c) s += Sl[tid * SS + c];
    rmean[tid] = s * (1.f / 128.f);
  }
  __syncthreads();
  if (tid < 128) {  // BN statistics over centered values
    float s = 0.f, sq = 0.f;
    for (int r = 0; r < 64; ++r) {
      float v = Sl[r * SS + tid] - rmean[r];
      s += v; sq += v * v;
    }
    atomicAdd(&ch_sum[tid], s);
    atomicAdd(&ch_sumsq[tid], sq);
  }
#pragma unroll
  for (int t = 0; t < 4; ++t) {
    int p = cg * 64 + t * 16 + l15;
#pragma unroll
    for (int r = 0; r < 8; ++r) {
      int row = m0 + r + hi8;
      out_cent[(((size_t)b * N + n0 + row) << 7) + p] = (bf16_t)(acc[t][r] - rmean[row]);
    }
  }
}

// stats buffer layout (floats): [0)qsum [128)qsumsq [256)ksum [384)ksumsq
//                               [512)qmean [640)qinv [768)kmean [896)kinv
__global__ void k_stats(float* __restrict__ st) {
  int t = threadIdx.x;
  if (t < 128) {
    float n = 8.f * 4096.f;
    float m = st[t] / n, v = st[128 + t] / n - m * m;
    st[512 + t] = m; st[640 + t] = rsqrtf(v + 1e-5f);
  } else {
    int c = t - 128;
    float n = 8.f * 1024.f;
    float m = st[256 + c] / n, v = st[384 + c] / n - m * m;
    st[768 + c] = m; st[896 + c] = rsqrtf(v + 1e-5f);
  }
}

__global__ void k_norm(const bf16_t* __restrict__ cent, bf16_t* __restrict__ outb,
                       const float* __restrict__ mean, const float* __restrict__ inv,
                       size_t total) {
  size_t i = (size_t)blockIdx.x * 256 + threadIdx.x;
  if (i >= total) return;
  int c = (int)(i & 127);
  outb[i] = (bf16_t)(((float)cent[i] - mean[c]) * inv[c]);
}

// global-context branch: mask softmax + out_gc[b][p]
__global__ void k_gc(const bf16_t* __restrict__ xp_t, const float* __restrict__ w_mask,
                     const float* __restrict__ b_mask, const float* __restrict__ v_f32,
                     float* __restrict__ gc) {
  __shared__ float mk[1024];
  __shared__ float red[256];
  int b = blockIdx.x, tid = threadIdx.x;
  const bf16_t* xb = xp_t + (size_t)b * 1024 * 256;
  for (int m = tid; m < 1024; m += 256) {
    float s = b_mask[0];
    const bf16_t* row = xb + (size_t)m * 256;
    for (int c = 0; c < 256; ++c) s += w_mask[c] * (float)row[c];
    mk[m] = s;
  }
  __syncthreads();
  float mx = -INFINITY;
  for (int m = tid; m < 1024; m += 256) mx = fmaxf(mx, mk[m]);
  red[tid] = mx; __syncthreads();
  for (int s = 128; s > 0; s >>= 1) {
    if (tid < s) red[tid] = fmaxf(red[tid], red[tid + s]);
    __syncthreads();
  }
  mx = red[0]; __syncthreads();
  float sm = 0.f;
  for (int m = tid; m < 1024; m += 256) { float e = __expf(mk[m] - mx); mk[m] = e; sm += e; }
  red[tid] = sm; __syncthreads();
  for (int s = 128; s > 0; s >>= 1) {
    if (tid < s) red[tid] += red[tid + s];
    __syncthreads();
  }
  float inv = 1.f / red[0];
  __syncthreads();
  if (tid < 128) {
    const float* vr = v_f32 + ((size_t)b * 128 + tid) * 1024;
    float s = 0.f;
    for (int m = 0; m < 1024; ++m) s += vr[m] * mk[m];
    gc[b * 128 + tid] = s * inv;
  }
}

// ---------------- fused flash attention (S=QtK, softmax, O=P Vt) ----------
__global__ void k_attn(const bf16_t* __restrict__ q_bf,  // [B][4096][128]
                       const bf16_t* __restrict__ k_bf,  // [B][1024][128]
                       const bf16_t* __restrict__ v_bf,  // [B][128][1024]
                       const float* __restrict__ gc,     // [B][128]
                       const float* __restrict__ gamma,
                       bf16_t* __restrict__ osim) {      // [B][4096][128]
  extern __shared__ char smem_dyn[];
  const int QS = 136, KS = 136, VS = 136, PS = 136, SS = 132;
  bf16_t* Ql = (bf16_t*)smem_dyn;   // [64][136]
  bf16_t* Kl = Ql + 64 * QS;        // [128][136]  (row nk: K[:,nk] over p)
  bf16_t* Vl = Kl + 128 * KS;       // [128][136]  (row p: V[p,:] over nk)
  bf16_t* Pl = Vl + 128 * VS;       // [64][136]
  float* Sl = (float*)(Pl + 64 * PS);  // [64][132]
  float* rowM = Sl + 64 * SS;
  float* rowL = rowM + 64;
  float* rowS = rowL + 64;
  int tid = threadIdx.x, lane = tid & 31, wid = tid >> 5;
  int b = blockIdx.y, nq0 = blockIdx.x * 64;
  int m0 = (wid >> 1) * 16, cg = wid & 1;
  int l15 = lane & 15, hi8 = (lane >> 4) * 8;
  const float inv_scale = 0.088388347648318447f;  // 1/sqrt(128)
  const bf16_t* qsrc = q_bf + (((size_t)b * 4096 + nq0) << 7);
  for (int i = tid; i < 64 * 16; i += 256) {          // Q: async stage
    int r = i >> 4, s = i & 15;
    cp_async_b128(&Ql[r * QS + s * 8], qsrc + ((size_t)r << 7) + s * 8);
  }
  if (tid < 64) { rowM[tid] = -INFINITY; rowL[tid] = 0.f; }
  v8f oacc[4] = {};
  for (int j = 0; j < 8; ++j) {
    __syncthreads();
    const bf16_t* ksrc = k_bf + (((size_t)b * 1024 + j * 128) << 7);
    for (int i = tid; i < 128 * 16; i += 256) {       // K chunk: async stage
      int r = i >> 4, s = i & 15;
      cp_async_b128(&Kl[r * KS + s * 8], ksrc + ((size_t)r << 7) + s * 8);
    }
    async_wait0();
    __syncthreads();
    v8f sacc[4] = {};
#pragma unroll
    for (int kk = 0; kk < 128; kk += 32) {   // K-dim = P = 128
      v16bf a = load_frag(&Ql[m0 * QS + kk], QS, lane);
#pragma unroll
      for (int t = 0; t < 4; ++t) {
        v16bf bm = load_frag(&Kl[(cg * 64 + t * 16) * KS + kk], KS, lane);
        sacc[t] = wmma_bf16(a, bm, sacc[t]);
      }
    }
#pragma unroll
    for (int t = 0; t < 4; ++t) {
      int c = cg * 64 + t * 16 + l15;
#pragma unroll
      for (int r = 0; r < 8; ++r)
        Sl[(m0 + r + hi8) * SS + c] = sacc[t][r] * inv_scale;
    }
    __syncthreads();
    if (tid < 64) {  // online softmax row update
      float mOld = rowM[tid], cm = -INFINITY;
      for (int c = 0; c < 128; ++c) cm = fmaxf(cm, Sl[tid * SS + c]);
      float mNew = fmaxf(mOld, cm);
      float corr = __expf(mOld - mNew);
      float s = 0.f;
      for (int c = 0; c < 128; ++c) {
        float e = __expf(Sl[tid * SS + c] - mNew);
        Pl[tid * PS + c] = (bf16_t)e;
        s += e;
      }
      rowL[tid] = rowL[tid] * corr + s;
      rowM[tid] = mNew;
      rowS[tid] = corr;
    }
    const bf16_t* vsrc = v_bf + ((size_t)b << 17) + j * 128;
    for (int i = tid; i < 128 * 16; i += 256) {       // V chunk: async stage
      int r = i >> 4, s = i & 15;
      cp_async_b128(&Vl[r * VS + s * 8], vsrc + ((size_t)r << 10) + s * 8);
    }
    async_wait0();
    __syncthreads();
#pragma unroll
    for (int r = 0; r < 8; ++r) {  // rescale running O
      float sc = rowS[m0 + r + hi8];
#pragma unroll
      for (int t = 0; t < 4; ++t) oacc[t][r] *= sc;
    }
#pragma unroll
    for (int kk = 0; kk < 128; kk += 32) {   // K-dim = chunk nk = 128
      v16bf a = load_frag(&Pl[m0 * PS + kk], PS, lane);
#pragma unroll
      for (int t = 0; t < 4; ++t) {
        v16bf bm = load_frag(&Vl[(cg * 64 + t * 16) * VS + kk], VS, lane);
        oacc[t] = wmma_bf16(a, bm, oacc[t]);
      }
    }
  }
  __syncthreads();
  float g = gamma[0];
#pragma unroll
  for (int t = 0; t < 4; ++t) {
    int p = cg * 64 + t * 16 + l15;
    float gcv = gc[b * 128 + p];
#pragma unroll
    for (int r = 0; r < 8; ++r) {
      int row = m0 + r + hi8;
      float val = oacc[t][r] / rowL[row] * g + gcv;
      osim[(((size_t)b * 4096 + nq0 + row) << 7) + p] = (bf16_t)val;
    }
  }
}

// ---------------- output GEMM: out = w_out * osim + x ----------------------
__global__ void k_out(const bf16_t* __restrict__ wout_bf,  // [256][128]
                      const bf16_t* __restrict__ osim,     // [B][4096][128]
                      const float* __restrict__ x,         // [B][256][4096]
                      float* __restrict__ out) {
  extern __shared__ char smem_dyn[];
  const int WSs = 136, OSs = 136;
  bf16_t* Wl = (bf16_t*)smem_dyn;  // [128 c][136]
  bf16_t* Ol = Wl + 128 * WSs;     // [128 n][136]
  int tid = threadIdx.x, lane = tid & 31, wid = tid >> 5;
  int b = blockIdx.z, c0 = blockIdx.y * 128, n0 = blockIdx.x * 128;
  int l15 = lane & 15, hi8 = (lane >> 4) * 8;
  const bf16_t* wsrc = wout_bf + ((size_t)c0 << 7);
  const bf16_t* osrc = osim + (((size_t)b * 4096 + n0) << 7);
  for (int i = tid; i < 128 * 16; i += 256) {
    int r = i >> 4, s = i & 15;
    cp_async_b128(&Wl[r * WSs + s * 8], wsrc + ((size_t)r << 7) + s * 8);
    cp_async_b128(&Ol[r * OSs + s * 8], osrc + ((size_t)r << 7) + s * 8);
  }
  async_wait0();
  __syncthreads();
  v8f acc[8] = {};
#pragma unroll
  for (int kk = 0; kk < 128; kk += 32) {
    v16bf a = load_frag(&Wl[(wid * 16) * WSs + kk], WSs, lane);
#pragma unroll
    for (int t = 0; t < 8; ++t) {
      v16bf bm = load_frag(&Ol[(t * 16) * OSs + kk], OSs, lane);
      acc[t] = wmma_bf16(a, bm, acc[t]);
    }
  }
  const float* xb = x + (((size_t)b * 256 + c0) << 12);
  float* ob = out + (((size_t)b * 256 + c0) << 12);
#pragma unroll
  for (int t = 0; t < 8; ++t) {
#pragma unroll
    for (int r = 0; r < 8; ++r) {
      int c = wid * 16 + r + hi8;
      int n = n0 + t * 16 + l15;
      size_t idx = ((size_t)c << 12) + n;
      ob[idx] = acc[t][r] + xb[idx];  // residual add
    }
  }
}

// ------------------------------- launcher ----------------------------------
extern "C" void kernel_launch(void* const* d_in, const int* in_sizes, int n_in,
                              void* d_out, int out_size, void* d_ws, size_t ws_size,
                              hipStream_t stream) {
  (void)in_sizes; (void)n_in; (void)out_size; (void)ws_size;
  const float* x      = (const float*)d_in[0];
  const float* w_q    = (const float*)d_in[1];
  const float* b_q    = (const float*)d_in[2];
  const float* w_k    = (const float*)d_in[3];
  const float* b_k    = (const float*)d_in[4];
  const float* w_v    = (const float*)d_in[5];
  const float* b_v    = (const float*)d_in[6];
  const float* w_out  = (const float*)d_in[7];
  const float* w_mask = (const float*)d_in[8];
  const float* b_mask = (const float*)d_in[9];
  const float* gamma  = (const float*)d_in[10];
  float* out = (float*)d_out;

  char* ws = (char*)d_ws;
  size_t off = 0;
  auto carve = [&](size_t bytes) -> void* {
    void* p = ws + off;
    off = (off + bytes + 255) & ~(size_t)255;
    return p;
  };
  bf16_t* x_t    = (bf16_t*)carve((size_t)8 * 4096 * 256 * 2);
  float*  xp     = (float*) carve((size_t)8 * 256 * 1024 * 4);
  bf16_t* xp_t   = (bf16_t*)carve((size_t)8 * 1024 * 256 * 2);
  bf16_t* wq_bf  = (bf16_t*)carve(128 * 256 * 2);
  bf16_t* wk_bf  = (bf16_t*)carve(128 * 256 * 2);
  bf16_t* wv_bf  = (bf16_t*)carve(128 * 256 * 2);
  bf16_t* wo_bf  = (bf16_t*)carve(256 * 128 * 2);
  bf16_t* q_cent = (bf16_t*)carve((size_t)8 * 4096 * 128 * 2);
  bf16_t* k_cent = (bf16_t*)carve((size_t)8 * 1024 * 128 * 2);
  bf16_t* q_bf   = (bf16_t*)carve((size_t)8 * 4096 * 128 * 2);
  bf16_t* k_bf   = (bf16_t*)carve((size_t)8 * 1024 * 128 * 2);
  bf16_t* v_bf   = (bf16_t*)carve((size_t)8 * 128 * 1024 * 2);
  float*  v_f32  = (float*) carve((size_t)8 * 128 * 1024 * 4);
  bf16_t* osim   = (bf16_t*)carve((size_t)8 * 4096 * 128 * 2);
  float*  stats  = (float*) carve(1024 * 4);
  float*  gc     = (float*) carve(8 * 128 * 4);

  hipMemsetAsync(stats, 0, 1024 * 4, stream);  // zero BN accumulators

  // weight casts
  k_cast<<<128, 256, 0, stream>>>(w_q, wq_bf, 128 * 256);
  k_cast<<<128, 256, 0, stream>>>(w_k, wk_bf, 128 * 256);
  k_cast<<<128, 256, 0, stream>>>(w_v, wv_bf, 128 * 256);
  k_cast<<<128, 256, 0, stream>>>(w_out, wo_bf, 256 * 128);

  // pool + transposes
  k_pool<<<(8 * 256 * 1024 + 255) / 256, 256, 0, stream>>>(x, xp);
  k_transpose_cast<<<dim3(4096 / 32, 256 / 32, 8), dim3(32, 8), 0, stream>>>(x, x_t, 256, 4096);
  k_transpose_cast<<<dim3(1024 / 32, 256 / 32, 8), dim3(32, 8), 0, stream>>>(xp, xp_t, 256, 1024);

  const size_t projShm = (size_t)64 * 72 * 2 + 128 * 72 * 2 + 64 * 132 * 4 + 64 * 4;
  k_proj<<<dim3(64, 8), 256, projShm, stream>>>(x_t, wq_bf, b_q, q_cent, nullptr, nullptr,
                                                stats + 0, stats + 128, 4096, 0);
  k_proj<<<dim3(16, 8), 256, projShm, stream>>>(xp_t, wk_bf, b_k, k_cent, nullptr, nullptr,
                                                stats + 256, stats + 384, 1024, 1);
  k_proj<<<dim3(16, 8), 256, projShm, stream>>>(xp_t, wv_bf, b_v, nullptr, v_bf, v_f32,
                                                nullptr, nullptr, 1024, 2);

  k_stats<<<1, 256, 0, stream>>>(stats);
  k_norm<<<(8 * 4096 * 128) / 256, 256, 0, stream>>>(q_cent, q_bf, stats + 512, stats + 640,
                                                     (size_t)8 * 4096 * 128);
  k_norm<<<(8 * 1024 * 128) / 256, 256, 0, stream>>>(k_cent, k_bf, stats + 768, stats + 896,
                                                     (size_t)8 * 1024 * 128);

  k_gc<<<8, 256, 0, stream>>>(xp_t, w_mask, b_mask, v_f32, gc);

  const size_t attnShm = (size_t)(64 * 136 * 2) + 128 * 136 * 2 + 128 * 136 * 2 +
                         64 * 136 * 2 + 64 * 132 * 4 + 3 * 64 * 4;
  k_attn<<<dim3(64, 8), 256, attnShm, stream>>>(q_bf, k_bf, v_bf, gc, gamma, osim);

  const size_t outShm = (size_t)2 * 128 * 136 * 2;
  k_out<<<dim3(32, 2, 8), 256, outShm, stream>>>(wo_bf, osim, x, out);
}
